// VSALocalAttention_31026843746326
// MI455X (gfx1250) — compile-verified, matrix operand order checked by reading, not run on hardware
//
#include <hip/hip_runtime.h>
#include <hip/hip_bf16.h>

// ---------------------------------------------------------------------------
// VSA local attention, MI455X (gfx1250), wave32 + WMMA bf16 pipeline,
// with GLOBAL_LOAD_ASYNC_TO_LDS staging for bf16 tiles.
// ---------------------------------------------------------------------------

typedef __attribute__((ext_vector_type(16))) __bf16 bf16x16;
typedef __attribute__((ext_vector_type(8)))  float  v8f;

#define B_    8
#define HH    128
#define WW    128
#define C_    256
#define HEADS 8
#define HD    32
#define WSZ   8
#define WN    16                 // windows per spatial dim
#define NWIN  (B_ * WN * WN)     // 2048 windows
#define NPIX  (B_ * HH * WW)     // 131072 pixels

// --- WMMA helpers ----------------------------------------------------------

__device__ __forceinline__ v8f wmma_bf16(bf16x16 a, bf16x16 b, v8f c) {
  // v_wmma_f32_16x16x32_bf16: (neg_a, A, neg_b, B, c_mod, C, reuse_a, reuse_b)
  return __builtin_amdgcn_wmma_f32_16x16x32_bf16(false, a, false, b, (short)0, c,
                                                 false, false);
}

// K-index pattern for 16-bit A/B operand fragments (ISA 7.12.2):
// VGPR0..3 hold K={0..7} (lanes 0-15) / K={8..15} (lanes 16-31),
// VGPR4..7 hold K={16..23} / K={24..31}; 2 elements per VGPR.
__device__ __forceinline__ int kpat(int e, int lane) {
  return ((e >> 1) & 3) * 2 + ((e & 8) ? 16 : 0) + ((lane & 16) ? 8 : 0);
}

// Load a 16x32 fragment where logical rows (M for A, N for Bt) are stored
// row-major with stride `stride` (elements). Works for LDS or global ptrs.
__device__ __forceinline__ bf16x16 frag_row(const __bf16* p, int stride) {
  const int lane = threadIdx.x & 31;
  const __bf16* row = p + (size_t)(lane & 15) * stride;
  bf16x16 f;
#pragma unroll
  for (int e = 0; e < 16; e += 2) {
    int k = kpat(e, lane);
    f[e]     = row[k];
    f[e + 1] = row[k + 1];
  }
  return f;
}

// --- Async global->LDS staging (CDNA5 TDM-adjacent path, ASYNCcnt) ---------
// Generic-pointer low 32 bits are the wave-relative LDS offset (ISA 10.2:
// flat addresses map to LDS by discarding the upper bits).

__device__ __forceinline__ void async_copy16(const __bf16* g, __bf16* l) {
  unsigned lo = (unsigned)(uintptr_t)l;
  unsigned long long ga = (unsigned long long)(uintptr_t)g;
  asm volatile("global_load_async_to_lds_b128 %0, %1, off"
               :: "v"(lo), "v"(ga) : "memory");
}

__device__ __forceinline__ void async_copy32(const __bf16* g, __bf16* l) {
  unsigned lo = (unsigned)(uintptr_t)l;
  unsigned long long ga = (unsigned long long)(uintptr_t)g;
  asm volatile("global_load_async_to_lds_b128 %0, %1, off\n\t"
               "global_load_async_to_lds_b128 %0, %1, off offset:16"
               :: "v"(lo), "v"(ga) : "memory");
}

__device__ __forceinline__ void async_wait0() {
  asm volatile("s_wait_asynccnt 0x0" ::: "memory");
}

// ---------------------------------------------------------------------------
// Kernel 1: QKV projection GEMM. C[M=NPIX][N=768] = x(f32) @ W_qkv + b_qkv,
// bf16 output. Block tile 64x64, 8 waves (2x4), K-steps of 32.
// (f32 sources need conversion, so staging stays on the VALU path here.)
// ---------------------------------------------------------------------------
__global__ void __launch_bounds__(256)
k_qkv_gemm(const float* __restrict__ x, const float* __restrict__ Wq,
           const float* __restrict__ bq, __bf16* __restrict__ qkv) {
  __shared__ __bf16 As[64 * 40];
  __shared__ __bf16 Bs[64 * 40];
  const int t = threadIdx.x;
  const int wave = t >> 5, lane = t & 31;
  const int n0 = blockIdx.x * 64;
  const int m0 = blockIdx.y * 64;
  const int wm = wave >> 2, wn = wave & 3;
  v8f acc0 = {0.f, 0.f, 0.f, 0.f, 0.f, 0.f, 0.f, 0.f};
  v8f acc1 = acc0;
  for (int k0 = 0; k0 < 256; k0 += 32) {
    {  // stage A 64x32 f32 -> bf16
      int i = t >> 2, kk = (t & 3) * 8;
      const float* src = x + (size_t)(m0 + i) * 256 + k0 + kk;
      __builtin_prefetch(src + 32, 0, 0);
#pragma unroll
      for (int u = 0; u < 8; ++u) As[i * 40 + kk + u] = (__bf16)src[u];
    }
    {  // stage Bt[n][k] = Wq[k0+k][n0+n]
      int k = t >> 3, nn = (t & 7) * 8;
      const float* src = Wq + (size_t)(k0 + k) * 768 + n0 + nn;
#pragma unroll
      for (int u = 0; u < 8; ++u) Bs[(nn + u) * 40 + k] = (__bf16)src[u];
    }
    __syncthreads();
    bf16x16 b  = frag_row(Bs + (wn * 16) * 40, 40);
    bf16x16 a0 = frag_row(As + (wm * 32) * 40, 40);
    bf16x16 a1 = frag_row(As + (wm * 32 + 16) * 40, 40);
    acc0 = wmma_bf16(a0, b, acc0);
    acc1 = wmma_bf16(a1, b, acc1);
    __syncthreads();
  }
  const int col = n0 + wn * 16 + (lane & 15);
  const float bias = bq[col];
  const int rbase = m0 + wm * 32 + ((lane & 16) ? 8 : 0);
#pragma unroll
  for (int v = 0; v < 8; ++v) {
    qkv[(size_t)(rbase + v) * 768 + col]      = (__bf16)(acc0[v] + bias);
    qkv[(size_t)(rbase + 16 + v) * 768 + col] = (__bf16)(acc1[v] + bias);
  }
}

// ---------------------------------------------------------------------------
// Kernel 2: channel-MLP gating. h = gelu(qkv @ mlp_w1 + b1); cw = softmax(
// h @ mlp_w2 + b2, heads). WMMA for the K=768,N=64 GEMM; VALU tail.
// A-tiles staged with async global->LDS copies (bf16 verbatim).
// ---------------------------------------------------------------------------
__global__ void __launch_bounds__(256)
k_gating(const __bf16* __restrict__ qkv, const float* __restrict__ w1,
         const float* __restrict__ b1, const float* __restrict__ w2,
         const float* __restrict__ b2, float* __restrict__ cw) {
  __shared__ __bf16 As[128 * 40];
  __shared__ __bf16 Bs[64 * 40];
  __shared__ float  tb[128 * 64];
  const int t = threadIdx.x;
  const int wave = t >> 5, lane = t & 31;
  const int m0 = blockIdx.x * 128;
  v8f acc[4];
#pragma unroll
  for (int nt = 0; nt < 4; ++nt)
    acc[nt] = (v8f){0.f, 0.f, 0.f, 0.f, 0.f, 0.f, 0.f, 0.f};
  for (int k0 = 0; k0 < 768; k0 += 32) {
    {  // async-stage A 128x32 bf16 (32B per thread, 16B-aligned both sides)
      int i = t >> 1, kk = (t & 1) * 16;
      async_copy32(qkv + (size_t)(m0 + i) * 768 + k0 + kk, As + i * 40 + kk);
    }
    {
      int k = t >> 3, nn = (t & 7) * 8;
      const float* src = w1 + (size_t)(k0 + k) * 64 + nn;
#pragma unroll
      for (int u = 0; u < 8; ++u) Bs[(nn + u) * 40 + k] = (__bf16)src[u];
    }
    async_wait0();
    __syncthreads();
    bf16x16 a = frag_row(As + (wave * 16) * 40, 40);
#pragma unroll
    for (int nt = 0; nt < 4; ++nt) {
      bf16x16 b = frag_row(Bs + (nt * 16) * 40, 40);
      acc[nt] = wmma_bf16(a, b, acc[nt]);
    }
    __syncthreads();
  }
  const int rrow = wave * 16 + ((lane & 16) ? 8 : 0);
#pragma unroll
  for (int nt = 0; nt < 4; ++nt) {
    int col = nt * 16 + (lane & 15);
    float bb = b1[col];
#pragma unroll
    for (int v = 0; v < 8; ++v) {
      float xv = acc[nt][v] + bb;
      tb[(rrow + v) * 64 + col] = 0.5f * xv * (1.0f + erff(xv * 0.7071067811865475f));
    }
  }
  __syncthreads();
  if (t < 128) {
    float h[8];
#pragma unroll
    for (int j = 0; j < 8; ++j) h[j] = b2[j];
    for (int kk = 0; kk < 64; ++kk) {
      float tv = tb[t * 64 + kk];
#pragma unroll
      for (int j = 0; j < 8; ++j) h[j] += tv * w2[kk * 8 + j];
    }
    float mx = h[0];
#pragma unroll
    for (int j = 1; j < 8; ++j) mx = fmaxf(mx, h[j]);
    float s = 0.f;
#pragma unroll
    for (int j = 0; j < 8; ++j) { h[j] = __expf(h[j] - mx); s += h[j]; }
    float inv = 1.f / s;
#pragma unroll
    for (int j = 0; j < 8; ++j) cw[(size_t)(m0 + t) * 8 + j] = h[j] * inv;
  }
}

// ---------------------------------------------------------------------------
// Kernel 3: sampling net (run twice: offsets, scales). Per-window fused
// conv3x3(q)+leaky -> 8x8 avg pool -> 1x1 conv. Implicit GEMM:
// M=64 window pixels, N=256, K=9*256, halo patch (10x10x256 bf16) staged
// into LDS with async copies (zero ds_stores for image-border padding).
// ---------------------------------------------------------------------------
__global__ void __launch_bounds__(256)
k_sampnet(const __bf16* __restrict__ qkv, const float* __restrict__ w1,
          const float* __restrict__ b1v, const float* __restrict__ w2,
          const float* __restrict__ b2v, float* __restrict__ outp) {
  __shared__ __bf16 patch[100 * 256];
  __shared__ __bf16 Bs[128 * 40];
  __shared__ float  pooled[256];
  const int t = threadIdx.x;
  const int wave = t >> 5, lane = t & 31;
  const int w = blockIdx.x;
  const int bi = w >> 8, wij = w & 255, wi = wij >> 4, wj = wij & 15;
  const int y0 = wi * 8 - 1, x0 = wj * 8 - 1;
  // Stage 10x10 halo patch of q (zero 'SAME' padding at image border).
  for (int i = t; i < 100 * 16; i += 256) {
    int pix = i >> 4, c0 = (i & 15) * 16;
    int py = pix / 10, px = pix % 10;
    int gy = y0 + py, gx = x0 + px;
    __bf16* dst = patch + pix * 256 + c0;
    if (gy >= 0 && gy < HH && gx >= 0 && gx < WW) {
      async_copy32(qkv + ((size_t)((bi * HH + gy) * WW + gx)) * 768 + c0, dst);
    } else {
#pragma unroll
      for (int u = 0; u < 16; ++u) dst[u] = (__bf16)0.0f;
    }
  }
  async_wait0();
  __syncthreads();
  for (int nh = 0; nh < 2; ++nh) {
    const int ncol0 = nh * 128;
    v8f acc[4];
#pragma unroll
    for (int mt = 0; mt < 4; ++mt)
      acc[mt] = (v8f){0.f, 0.f, 0.f, 0.f, 0.f, 0.f, 0.f, 0.f};
    for (int ks = 0; ks < 72; ++ks) {
      const int tap = ks >> 3, kc = (ks & 7) * 32;
      const int dy = tap / 3, dx = tap % 3;
      {  // stage Bt[n][k] = w1[dy][dx][kc+k][ncol0+n]
        int k = t >> 3, nn = (t & 7) * 16;
        const float* src = w1 + ((size_t)tap * 256 + kc + k) * 256 + ncol0 + nn;
#pragma unroll
        for (int u = 0; u < 16; ++u) Bs[(nn + u) * 40 + k] = (__bf16)src[u];
      }
      __syncthreads();
      bf16x16 b = frag_row(Bs + (wave * 16) * 40, 40);
#pragma unroll
      for (int mt = 0; mt < 4; ++mt) {
        int m = mt * 16 + (lane & 15);
        int wy = m >> 3, wx = m & 7;
        const __bf16* arow = patch + ((wy + dy) * 10 + (wx + dx)) * 256 + kc;
        bf16x16 a;
#pragma unroll
        for (int e = 0; e < 16; e += 2) {
          int k = kpat(e, lane);
          a[e]     = arow[k];
          a[e + 1] = arow[k + 1];
        }
        acc[mt] = wmma_bf16(a, b, acc[mt]);
      }
      __syncthreads();
    }
    // leaky_relu then 8x8 average pool (sum rows, /64).
    const float bb = b1v[ncol0 + wave * 16 + (lane & 15)];
    float s = 0.f;
#pragma unroll
    for (int mt = 0; mt < 4; ++mt)
#pragma unroll
      for (int v = 0; v < 8; ++v) {
        float xv = acc[mt][v] + bb;
        s += (xv >= 0.f) ? xv : 0.01f * xv;
      }
    s += __shfl_xor(s, 16, 32);  // merge the two lane-half row sets
    if (lane < 16) pooled[ncol0 + wave * 16 + lane] = s * (1.0f / 64.0f);
  }
  __syncthreads();
  if (t < 16) {  // 1x1 conv 256 -> 16 (2*heads)
    float o = b2v[t];
    for (int c = 0; c < 256; ++c) o += pooled[c] * w2[c * 16 + t];
    outp[(size_t)w * 16 + t] = o;
  }
}

// ---------------------------------------------------------------------------
// Kernel 4: per-(window,head) deformable bilinear sampling of k and v into
// WMMA-ready bf16 tiles [win][head][token][hd].
// ---------------------------------------------------------------------------
__global__ void __launch_bounds__(256)
k_sample(const __bf16* __restrict__ qkv, const float* __restrict__ offp,
         const float* __restrict__ scp, __bf16* __restrict__ kw,
         __bf16* __restrict__ vw) {
  const int w = blockIdx.x;
  const int bi = w >> 8, wij = w & 255, wi = wij >> 4, wj = wij & 15;
  const int t = threadIdx.x;
  for (int pair = t; pair < 512; pair += 256) {
    const int h = pair >> 6, tok = pair & 63;
    const int ty = tok >> 3, tx = tok & 7;
    const float ofx = 0.3f * tanhf(offp[w * 16 + h * 2 + 0]);
    const float ofy = 0.3f * tanhf(offp[w * 16 + h * 2 + 1]);
    const float sfx = 1.0f + 0.5f * tanhf(scp[w * 16 + h * 2 + 0]);
    const float sfy = 1.0f + 0.5f * tanhf(scp[w * 16 + h * 2 + 1]);
    const float bgx = -1.0f + (2.0f / 7.0f) * (float)tx;
    const float bgy = -1.0f + (2.0f / 7.0f) * (float)ty;
    const float gx = fminf(fmaxf(bgx * sfx + ofx, -1.f), 1.f);
    const float gy = fminf(fmaxf(bgy * sfy + ofy, -1.f), 1.f);
    const float ix = ((gx + 1.f) * 8.f - 1.f) * 0.5f;
    const float iy = ((gy + 1.f) * 8.f - 1.f) * 0.5f;
    const float x0f = floorf(ix), y0f = floorf(iy);
    const int xx0 = (int)x0f, yy0 = (int)y0f;
    const float wx1 = ix - x0f, wx0 = 1.f - wx1;
    const float wy1 = iy - y0f, wy0 = 1.f - wy1;
    const int   xs[2]  = {xx0, xx0 + 1}, ys[2] = {yy0, yy0 + 1};
    const float wxs[2] = {wx0, wx1},     wys[2] = {wy0, wy1};
    const size_t chK = 256 + (size_t)h * 32;  // k channels in qkv row
    const size_t outbase = (((size_t)w * 8 + h) * 64 + tok) * 32;
    float aK[32], aV[32];
#pragma unroll
    for (int c = 0; c < 32; ++c) { aK[c] = 0.f; aV[c] = 0.f; }
#pragma unroll
    for (int cy = 0; cy < 2; ++cy)
#pragma unroll
      for (int cx = 0; cx < 2; ++cx) {
        const int yy = ys[cy], xx = xs[cx];
        const float wt = wys[cy] * wxs[cx];
        if (yy >= 0 && yy < 8 && xx >= 0 && xx < 8) {
          const size_t p =
              (size_t)(bi * HH + wi * 8 + yy) * WW + (wj * 8 + xx);
          const __bf16* kp = qkv + p * 768 + chK;
#pragma unroll
          for (int c = 0; c < 32; ++c) {
            aK[c] += wt * (float)kp[c];
            aV[c] += wt * (float)kp[256 + c];
          }
        }
      }
#pragma unroll
    for (int c = 0; c < 32; ++c) {
      kw[outbase + c] = (__bf16)aK[c];
      vw[outbase + c] = (__bf16)aV[c];
    }
  }
}

// ---------------------------------------------------------------------------
// Kernel 5: windowed attention. One wave per (window, head): QK^T (16 WMMA),
// rel-pos bias from LDS, in-register softmax (16-lane shuffles), P staged
// through LDS to reshape into A-operand layout, P@V (16 WMMA), cw gating.
// Block = 4 waves; grid = 2 blocks per window.
// ---------------------------------------------------------------------------
__global__ void __launch_bounds__(128)
k_attn(const __bf16* __restrict__ qkv, const __bf16* __restrict__ kw,
       const __bf16* __restrict__ vw, const float* __restrict__ rpb,
       const float* __restrict__ cw, __bf16* __restrict__ aout) {
  __shared__ float  rpbs[225 * 8];
  __shared__ __bf16 pbuf[4][64 * 72];
  const int t = threadIdx.x;
  const int wave = t >> 5, lane = t & 31, kx = lane & 15;
  const int blk = blockIdx.x;
  const int w = blk >> 1;
  const int h = (blk & 1) * 4 + wave;
  const int bi = w >> 8, wij = w & 255, wi = wij >> 4, wj = wij & 15;
  for (int i = t; i < 225 * 8; i += 128) rpbs[i] = rpb[i];
  __syncthreads();
  const size_t prow0 = ((size_t)(bi * HH + wi * 8) * WW + wj * 8);
  // Q fragments (A operand), scale folded into logits later.
  bf16x16 aq[4];
#pragma unroll
  for (int mt = 0; mt < 4; ++mt) {
    int m = mt * 16 + kx;
    int ty = m >> 3, tx = m & 7;
    const __bf16* row = qkv + (prow0 + (size_t)ty * WW + tx) * 768 + h * 32;
    bf16x16 a;
#pragma unroll
    for (int e = 0; e < 16; e += 2) {
      int k = kpat(e, lane);
      a[e]     = row[k];
      a[e + 1] = row[k + 1];
    }
    aq[mt] = a;
  }
  const __bf16* kbase = kw + (((size_t)w * 8 + h) * 64) * 32;
  const __bf16* vbase = vw + (((size_t)w * 8 + h) * 64) * 32;
  v8f s[4][4];
#pragma unroll
  for (int mt = 0; mt < 4; ++mt)
#pragma unroll
    for (int nt = 0; nt < 4; ++nt)
      s[mt][nt] = (v8f){0.f, 0.f, 0.f, 0.f, 0.f, 0.f, 0.f, 0.f};
#pragma unroll
  for (int nt = 0; nt < 4; ++nt) {
    bf16x16 bk = frag_row(kbase + (size_t)(nt * 16) * 32, 32);
#pragma unroll
    for (int mt = 0; mt < 4; ++mt) s[mt][nt] = wmma_bf16(aq[mt], bk, s[mt][nt]);
  }
  const float scale = 0.17677669529663687f;  // 32^-0.5
#pragma unroll
  for (int mt = 0; mt < 4; ++mt) {
#pragma unroll
    for (int v = 0; v < 8; ++v) {
      const int qtok = mt * 16 + v + ((lane & 16) ? 8 : 0);
      const int qy = qtok >> 3, qx = qtok & 7;
      float mx = -1e30f;
#pragma unroll
      for (int nt = 0; nt < 4; ++nt) {
        const int ktok = nt * 16 + kx;
        const int ky = ktok >> 3, kxx = ktok & 7;
        const int idx = (qy - ky + 7) * 15 + (qx - kxx + 7);
        float val = s[mt][nt][v] * scale + rpbs[idx * 8 + h];
        s[mt][nt][v] = val;
        mx = fmaxf(mx, val);
      }
#pragma unroll
      for (int off = 1; off < 16; off <<= 1)
        mx = fmaxf(mx, __shfl_xor(mx, off, 32));
      float sum = 0.f;
#pragma unroll
      for (int nt = 0; nt < 4; ++nt) {
        float e = __expf(s[mt][nt][v] - mx);
        s[mt][nt][v] = e;
        sum += e;
      }
#pragma unroll
      for (int off = 1; off < 16; off <<= 1) sum += __shfl_xor(sum, off, 32);
      const float inv = 1.f / sum;
#pragma unroll
      for (int nt = 0; nt < 4; ++nt) s[mt][nt][v] *= inv;
    }
  }
  // Stage P (D-layout) into wave-private LDS so it can be reloaded in A-layout.
  __bf16* pb = pbuf[wave];
#pragma unroll
  for (int mt = 0; mt < 4; ++mt)
#pragma unroll
    for (int nt = 0; nt < 4; ++nt) {
      const int q0 = mt * 16 + ((lane & 16) ? 8 : 0);
      const int ktok = nt * 16 + kx;
#pragma unroll
      for (int v = 0; v < 8; ++v)
        pb[(q0 + v) * 72 + ktok] = (__bf16)s[mt][nt][v];
    }
  // P @ V : K = 64 tokens in two 32-chunks; N = 32 head-dim in two 16-tiles.
  v8f o[4][2];
#pragma unroll
  for (int mt = 0; mt < 4; ++mt)
#pragma unroll
    for (int nt = 0; nt < 2; ++nt)
      o[mt][nt] = (v8f){0.f, 0.f, 0.f, 0.f, 0.f, 0.f, 0.f, 0.f};
#pragma unroll
  for (int kc = 0; kc < 2; ++kc) {
    bf16x16 ap[4];
#pragma unroll
    for (int mt = 0; mt < 4; ++mt)
      ap[mt] = frag_row(pb + (size_t)(mt * 16) * 72 + kc * 32, 72);
#pragma unroll
    for (int nt = 0; nt < 2; ++nt) {
      // B fragment: Bt[n=ch][k=tok] from vbase[tok][ch] (transposed gather).
      const __bf16* col = vbase + (size_t)kc * 32 * 32 + nt * 16 + kx;
      bf16x16 bv;
#pragma unroll
      for (int e = 0; e < 16; e += 2) {
        int k = kpat(e, lane);
        bv[e]     = col[(size_t)k * 32];
        bv[e + 1] = col[(size_t)(k + 1) * 32];
      }
#pragma unroll
      for (int mt = 0; mt < 4; ++mt) o[mt][nt] = wmma_bf16(ap[mt], bv, o[mt][nt]);
    }
  }
  // Gate by cw and store bf16 for the projection GEMM.
#pragma unroll
  for (int mt = 0; mt < 4; ++mt) {
    const int q0 = mt * 16 + ((lane & 16) ? 8 : 0);
#pragma unroll
    for (int v = 0; v < 8; ++v) {
      const int qtok = q0 + v;
      const int ty = qtok >> 3, tx = qtok & 7;
      const size_t p = prow0 + (size_t)ty * WW + tx;
      const float g = cw[p * 8 + h];
#pragma unroll
      for (int nt = 0; nt < 2; ++nt) {
        const int ch = nt * 16 + kx;
        aout[p * 256 + h * 32 + ch] = (__bf16)(o[mt][nt][v] * g);
      }
    }
  }
}

// ---------------------------------------------------------------------------
// Kernel 6: output projection GEMM. out[M=NPIX][256] = aout(bf16) @ proj_w
// + proj_b, fp32 output to d_out. A-tiles staged via async global->LDS.
// ---------------------------------------------------------------------------
__global__ void __launch_bounds__(256)
k_proj(const __bf16* __restrict__ ain, const float* __restrict__ Wp,
       const float* __restrict__ bp, float* __restrict__ out) {
  __shared__ __bf16 As[64 * 40];
  __shared__ __bf16 Bs[64 * 40];
  const int t = threadIdx.x;
  const int wave = t >> 5, lane = t & 31;
  const int n0 = blockIdx.x * 64;
  const int m0 = blockIdx.y * 64;
  const int wm = wave >> 2, wn = wave & 3;
  v8f acc0 = {0.f, 0.f, 0.f, 0.f, 0.f, 0.f, 0.f, 0.f};
  v8f acc1 = acc0;
  for (int k0 = 0; k0 < 256; k0 += 32) {
    {  // async-stage A 64x32 bf16 (16B per thread, aligned both sides)
      int i = t >> 2, kk = (t & 3) * 8;
      async_copy16(ain + (size_t)(m0 + i) * 256 + k0 + kk, As + i * 40 + kk);
    }
    {
      int k = t >> 3, nn = (t & 7) * 8;
      const float* src = Wp + (size_t)(k0 + k) * 256 + n0 + nn;
#pragma unroll
      for (int u = 0; u < 8; ++u) Bs[(nn + u) * 40 + k] = (__bf16)src[u];
    }
    async_wait0();
    __syncthreads();
    bf16x16 b  = frag_row(Bs + (wn * 16) * 40, 40);
    bf16x16 a0 = frag_row(As + (wm * 32) * 40, 40);
    bf16x16 a1 = frag_row(As + (wm * 32 + 16) * 40, 40);
    acc0 = wmma_bf16(a0, b, acc0);
    acc1 = wmma_bf16(a1, b, acc1);
    __syncthreads();
  }
  const int col = n0 + wn * 16 + (lane & 15);
  const float bias = bp[col];
  const int rbase = m0 + wm * 32 + ((lane & 16) ? 8 : 0);
#pragma unroll
  for (int v = 0; v < 8; ++v) {
    out[(size_t)(rbase + v) * 256 + col]      = acc0[v] + bias;
    out[(size_t)(rbase + 16 + v) * 256 + col] = acc1[v] + bias;
  }
}

// ---------------------------------------------------------------------------
extern "C" void kernel_launch(void* const* d_in, const int* in_sizes, int n_in,
                              void* d_out, int out_size, void* d_ws,
                              size_t ws_size, hipStream_t stream) {
  const float* x      = (const float*)d_in[0];
  const float* W_qkv  = (const float*)d_in[1];
  const float* b_qkv  = (const float*)d_in[2];
  const float* off_w1 = (const float*)d_in[3];
  const float* off_b1 = (const float*)d_in[4];
  const float* off_w2 = (const float*)d_in[5];
  const float* off_b2 = (const float*)d_in[6];
  const float* sc_w1  = (const float*)d_in[7];
  const float* sc_b1  = (const float*)d_in[8];
  const float* sc_w2  = (const float*)d_in[9];
  const float* sc_b2  = (const float*)d_in[10];
  const float* rpb    = (const float*)d_in[11];
  const float* mlp_w1 = (const float*)d_in[12];
  const float* mlp_b1 = (const float*)d_in[13];
  const float* mlp_w2 = (const float*)d_in[14];
  const float* mlp_b2 = (const float*)d_in[15];
  const float* proj_w = (const float*)d_in[16];
  const float* proj_b = (const float*)d_in[17];

  char* ws = (char*)d_ws;
  __bf16* qkv = (__bf16*)ws;           ws += (size_t)NPIX * 768 * 2;
  float*  cwb = (float*)ws;            ws += (size_t)NPIX * 8 * 4;
  float*  offo = (float*)ws;           ws += (size_t)NWIN * 16 * 4;
  float*  sco  = (float*)ws;           ws += (size_t)NWIN * 16 * 4;
  __bf16* kwb = (__bf16*)ws;           ws += (size_t)NWIN * 8 * 64 * 32 * 2;
  __bf16* vwb = (__bf16*)ws;           ws += (size_t)NWIN * 8 * 64 * 32 * 2;
  __bf16* aoutb = (__bf16*)ws;         ws += (size_t)NPIX * 256 * 2;

  k_qkv_gemm<<<dim3(12, NPIX / 64), 256, 0, stream>>>(x, W_qkv, b_qkv, qkv);
  k_gating<<<NPIX / 128, 256, 0, stream>>>(qkv, mlp_w1, mlp_b1, mlp_w2, mlp_b2,
                                           cwb);
  k_sampnet<<<NWIN, 256, 0, stream>>>(qkv, off_w1, off_b1, off_w2, off_b2, offo);
  k_sampnet<<<NWIN, 256, 0, stream>>>(qkv, sc_w1, sc_b1, sc_w2, sc_b2, sco);
  k_sample<<<NWIN, 256, 0, stream>>>(qkv, offo, sco, kwb, vwb);
  k_attn<<<NWIN * 2, 128, 0, stream>>>(qkv, kwb, vwb, rpb, cwb, aoutb);
  k_proj<<<dim3(4, NPIX / 64), 256, 0, stream>>>(aoutb, proj_w, proj_b,
                                                 (float*)d_out);
}
